// Multi_Head_Attention_678604833534
// MI455X (gfx1250) — matrix operand order repaired and checked by reading, not compiled, not run on hardware
//
#include <hip/hip_runtime.h>
#include <stdint.h>

// Problem constants (match reference)
#define B_  2
#define S_  2048
#define D_  1024
#define H_  16
#define DK_ 64
#define M_  (B_ * S_)   // 4096 total rows

// Padded LDS row strides (break stride-2^k bank patterns, keep 16B alignment)
#define LD64 72   // for 64-wide tiles  (72*2B = 144B rows)
#define LD32 40   // for 32-wide tiles  (40*2B = 80B rows)

typedef __bf16 bf16_t;
typedef __attribute__((ext_vector_type(16))) __bf16 v16bf;
typedef __attribute__((ext_vector_type(8)))  __bf16 v8bf;
typedef __attribute__((ext_vector_type(8)))  float  v8f;

// f32 -> bf16 round-to-nearest-even (scalar fallback)
static __device__ __forceinline__ bf16_t f2bf(float f) {
  union { float f; uint32_t u; } c; c.f = f;
  uint32_t u = c.u;
  uint32_t r = (u + 0x7FFFu + ((u >> 16) & 1u)) >> 16;
  union { uint16_t u; bf16_t b; } o; o.u = (uint16_t)r;
  return o.b;
}

// Pack two f32 -> two bf16 in one u32 (v_cvt_pk_bf16_f32 when available)
#if __has_builtin(__builtin_amdgcn_cvt_pk_bf16_f32)
static __device__ __forceinline__ uint32_t f2bf_pk(float a, float b) {
  auto v = __builtin_amdgcn_cvt_pk_bf16_f32(a, b);
  return __builtin_bit_cast(uint32_t, v);
}
#else
static __device__ __forceinline__ uint32_t f2bf_pk(float a, float b) {
  union { uint16_t u; bf16_t t; } x, y; x.t = f2bf(a); y.t = f2bf(b);
  return (uint32_t)x.u | ((uint32_t)y.u << 16);
}
#endif

// Async DMA: 16 bytes per lane, global -> LDS, tracked by ASYNCcnt.
static __device__ __forceinline__ void async_b128(const void* gaddr, void* lds) {
  uint32_t l = (uint32_t)(uintptr_t)lds;           // LDS byte address = addr[31:0]
  uint64_t g = (uint64_t)(uintptr_t)gaddr;
  asm volatile("global_load_async_to_lds_b128 %0, %1, off"
               :: "v"(l), "v"(g) : "memory");
}
static __device__ __forceinline__ void wait_async0() {
  asm volatile("s_wait_asynccnt 0" ::: "memory");
}

// B-matrix fragment: 16 contiguous bf16 (one column's K-run)
static __device__ __forceinline__ v16bf ld_frag_b16(const bf16_t* p) {
  union { v16bf v; v8bf h[2]; } u;
  u.h[0] = *(const v8bf*)(p);
  u.h[1] = *(const v8bf*)(p + 8);
  return u.v;
}
// A-matrix fragment: 16 contiguous bf16 (one row's K-run)
static __device__ __forceinline__ v16bf ld_frag_a16(const bf16_t* p) {
  union { v16bf v; v8bf h[2]; } u;
  u.h[0] = *(const v8bf*)(p);
  u.h[1] = *(const v8bf*)(p + 8);
  return u.v;
}

#define WMMA_BF16(a, b, c) \
  __builtin_amdgcn_wmma_f32_16x16x32_bf16(false, (a), false, (b), (short)0, (c), false, false)

// ---------------------------------------------------------------------------
// Kernel 1: per-head projection  Y[b,h,s,:] = X[b,s,:] @ W[h] + bias[h]
// GEMM M=4096, N=64, K=1024. Block = 256 thr (8 waves), tile 128x64, Kstep 32.
// ---------------------------------------------------------------------------
__global__ __launch_bounds__(256)
void qkv_proj_kernel(const float* __restrict__ X, const float* __restrict__ W,
                     const float* __restrict__ bias, bf16_t* __restrict__ Y) {
  __shared__ bf16_t As[128 * LD32];  // [row][k], padded
  __shared__ bf16_t Bs[64 * LD32];   // [n][k] transposed, padded
  const int tid  = threadIdx.x;
  const int wave = tid >> 5, lane = tid & 31;
  const int hi   = lane >> 4, lc = lane & 15;
  const int m0   = blockIdx.x * 128;
  const int h    = blockIdx.y;

  v8f acc[4] = {};
  for (int k0 = 0; k0 < D_; k0 += 32) {
    // stage A: 128x32 f32 -> bf16, packed pair stores (2048 u32)
#pragma unroll
    for (int i = 0; i < 8; ++i) {
      int idx = tid + i * 256;
      int r = idx >> 4, w = idx & 15;
      const float* src = X + (size_t)(m0 + r) * D_ + k0 + 2 * w;
      ((uint32_t*)As)[r * (LD32 / 2) + w] = f2bf_pk(src[0], src[1]);
    }
    // stage B transposed: W[k][n] -> Bs[n][k], packed pairs along k (1024 u32)
#pragma unroll
    for (int i = 0; i < 4; ++i) {
      int idx = tid + i * 256;
      int n = idx & 63, kp = idx >> 6;
      const float* src = W + ((size_t)h * D_ + k0 + 2 * kp) * DK_ + n;
      ((uint32_t*)Bs)[n * (LD32 / 2) + kp] = f2bf_pk(src[0], src[DK_]);
    }
    if (k0 + 32 < D_)
      __builtin_prefetch(X + (size_t)(m0 + (tid >> 1)) * D_ + k0 + 32, 0, 1);
    __syncthreads();
    v16bf a = ld_frag_a16(&As[(wave * 16 + lc) * LD32 + hi * 16]);
#pragma unroll
    for (int nt = 0; nt < 4; ++nt) {
      v16bf b = ld_frag_b16(&Bs[(nt * 16 + lc) * LD32 + hi * 16]);
      acc[nt] = WMMA_BF16(a, b, acc[nt]);
    }
    __syncthreads();
  }
#pragma unroll
  for (int nt = 0; nt < 4; ++nt) {
#pragma unroll
    for (int r = 0; r < 8; ++r) {
      int row = wave * 16 + hi * 8 + r;
      int col = nt * 16 + lc;
      int gm  = m0 + row;
      int bb  = gm >> 11, s = gm & (S_ - 1);
      float v = acc[nt][r] + bias[h * DK_ + col];
      Y[(((size_t)bb * H_ + h) * S_ + s) * DK_ + col] = f2bf(v);
    }
  }
}

// ---------------------------------------------------------------------------
// Kernel 2: attention. One block = (b, h, 128-row q tile). Q staged by async
// DMA; K tiles double-buffered via GLOBAL_LOAD_ASYNC_TO_LDS_B128 so the next
// tile's HBM/L2 traffic overlaps current-tile WMMA. Single-pass softmax.
// ---------------------------------------------------------------------------
__global__ __launch_bounds__(256)
void attn_kernel(const bf16_t* __restrict__ qh, const bf16_t* __restrict__ kh,
                 const bf16_t* __restrict__ vh, const int* __restrict__ mask,
                 bf16_t* __restrict__ concat, float* __restrict__ attn_out,
                 float* __restrict__ lsum) {
  __shared__ bf16_t Qs[128 * LD64];        // [row][dk]
  __shared__ bf16_t Ks[2][64 * LD64];      // [key][dk], double buffered (async)
  __shared__ bf16_t Vt[2][64 * LD64];      // [dv][key], double buffered (manual T)
  __shared__ bf16_t Ps[8][16 * LD64];      // per-wave P tile

  const int tid  = threadIdx.x;
  const int wave = tid >> 5, lane = tid & 31;
  const int hi   = lane >> 4, lc = lane & 15;
  const int q0   = blockIdx.x * 128;
  const int h    = blockIdx.y;
  const int b    = blockIdx.z;
  const bool last_head = (h == H_ - 1);
  const size_t base = ((size_t)b * H_ + h) * (size_t)S_ * DK_;
  const int NT = S_ / 64;

  // ---- prologue: async Q tile (1024 chunks), async K[0], manual Vt[0] ----
#pragma unroll
  for (int i = 0; i < 4; ++i) {           // Q: 128 rows x 64 dk = 1024 x 16B
    int idx = tid + i * 256;
    int r = idx >> 3, seg = idx & 7;
    async_b128(qh + base + (size_t)(q0 + r) * DK_ + seg * 8, &Qs[r * LD64 + seg * 8]);
  }
#pragma unroll
  for (int i = 0; i < 2; ++i) {           // K tile 0: 64 x 64 = 512 x 16B
    int idx = tid + i * 256;
    int key = idx >> 3, seg = idx & 7;
    async_b128(kh + base + (size_t)key * DK_ + seg * 8, &Ks[0][key * LD64 + seg * 8]);
  }
#pragma unroll
  for (int i = 0; i < 8; ++i) {           // Vt[0]: packed pair stores (2048 u32)
    int idx = tid + i * 256;
    int w = idx >> 6, dv = idx & 63;      // keys 2w, 2w+1
    uint32_t lo = *(const uint16_t*)(vh + base + (size_t)(2 * w) * DK_ + dv);
    uint32_t hv = *(const uint16_t*)(vh + base + (size_t)(2 * w + 1) * DK_ + dv);
    ((uint32_t*)&Vt[0][dv * LD64])[w] = lo | (hv << 16);
  }
  wait_async0();
  __syncthreads();

  v8f   o[4] = {};
  float lpart[8] = {0, 0, 0, 0, 0, 0, 0, 0};

  for (int kt = 0; kt < NT; ++kt) {
    const int k0  = kt * 64;
    const int cur = kt & 1, nxt = cur ^ 1;

    if (kt + 1 < NT) {  // kick off next tile: async K, manual Vt
      const size_t nb = base + (size_t)(k0 + 64) * DK_;
#pragma unroll
      for (int i = 0; i < 2; ++i) {
        int idx = tid + i * 256;
        int key = idx >> 3, seg = idx & 7;
        async_b128(kh + nb + (size_t)key * DK_ + seg * 8, &Ks[nxt][key * LD64 + seg * 8]);
      }
#pragma unroll
      for (int i = 0; i < 8; ++i) {
        int idx = tid + i * 256;
        int w = idx >> 6, dv = idx & 63;
        uint32_t lo = *(const uint16_t*)(vh + nb + (size_t)(2 * w) * DK_ + dv);
        uint32_t hv = *(const uint16_t*)(vh + nb + (size_t)(2 * w + 1) * DK_ + dv);
        ((uint32_t*)&Vt[nxt][dv * LD64])[w] = lo | (hv << 16);
      }
    }

    // ---- scores = Q @ K^T  (DK = 64 -> two bf16 WMMAs per 16x16 tile) ----
    v8f sc[4] = {};
    v16bf a0 = ld_frag_a16(&Qs[(wave * 16 + lc) * LD64 + 0  + hi * 16]);
    v16bf a1 = ld_frag_a16(&Qs[(wave * 16 + lc) * LD64 + 32 + hi * 16]);
#pragma unroll
    for (int nt = 0; nt < 4; ++nt) {
      v16bf bk0 = ld_frag_b16(&Ks[cur][(nt * 16 + lc) * LD64 + 0  + hi * 16]);
      v16bf bk1 = ld_frag_b16(&Ks[cur][(nt * 16 + lc) * LD64 + 32 + hi * 16]);
      sc[nt] = WMMA_BF16(a0, bk0, sc[nt]);
      sc[nt] = WMMA_BF16(a1, bk1, sc[nt]);
    }

    // ---- exp, row-sum accumulate, write P to per-wave LDS ----
#pragma unroll
    for (int nt = 0; nt < 4; ++nt) {
#pragma unroll
      for (int r = 0; r < 8; ++r) {
        int lrow = hi * 8 + r;
        int row  = wave * 16 + lrow;
        int key  = nt * 16 + lc;
        int mv   = mask[((size_t)b * S_ + (q0 + row)) * S_ + (k0 + key)];
        float e  = __expf(sc[nt][r] * 0.125f + (float)mv * -1e9f);
        lpart[r] += e;
        Ps[wave][lrow * LD64 + key] = f2bf(e);
        if (last_head)
          attn_out[((size_t)b * S_ + (q0 + row)) * S_ + (k0 + key)] = e;
      }
    }
    // own-wave LDS RAW across lanes: drain DS counter before A-frag reads
    asm volatile("s_wait_dscnt 0" ::: "memory");

    // ---- O += P @ V ----
    v16bf pa0 = ld_frag_a16(&Ps[wave][lc * LD64 + 0  + hi * 16]);
    v16bf pa1 = ld_frag_a16(&Ps[wave][lc * LD64 + 32 + hi * 16]);
#pragma unroll
    for (int nt = 0; nt < 4; ++nt) {
      v16bf vb0 = ld_frag_b16(&Vt[cur][(nt * 16 + lc) * LD64 + 0  + hi * 16]);
      v16bf vb1 = ld_frag_b16(&Vt[cur][(nt * 16 + lc) * LD64 + 32 + hi * 16]);
      o[nt] = WMMA_BF16(pa0, vb0, o[nt]);
      o[nt] = WMMA_BF16(pa1, vb1, o[nt]);
    }
    wait_async0();        // next K tile landed
    __syncthreads();      // all waves done with cur + staging visible
  }

  // ---- finish row sums: reduce across the 16 lanes of each wave half ----
#pragma unroll
  for (int r = 0; r < 8; ++r) {
    float v = lpart[r];
    v += __shfl_xor(v, 1);
    v += __shfl_xor(v, 2);
    v += __shfl_xor(v, 4);
    v += __shfl_xor(v, 8);
    lpart[r] = v;  // full row sum for row wave*16 + hi*8 + r
  }
  if (last_head && lc == 0) {
#pragma unroll
    for (int r = 0; r < 8; ++r)
      lsum[(size_t)b * S_ + q0 + wave * 16 + hi * 8 + r] = lpart[r];
  }

  // ---- normalize and store concat (bf16) ----
#pragma unroll
  for (int nt = 0; nt < 4; ++nt) {
#pragma unroll
    for (int r = 0; r < 8; ++r) {
      int row = wave * 16 + hi * 8 + r;
      int col = nt * 16 + lc;
      float v = o[nt][r] * (1.0f / lpart[r]);
      concat[((size_t)b * S_ + (q0 + row)) * D_ + h * DK_ + col] = f2bf(v);
    }
  }
}

// ---------------------------------------------------------------------------
// Kernel 3: normalize last-head attention weights: attn /= rowsum
// ---------------------------------------------------------------------------
__global__ __launch_bounds__(256)
void attn_norm_kernel(float* __restrict__ attn_out, const float* __restrict__ lsum) {
  int gid = blockIdx.x * 256 + threadIdx.x;   // 4 elems / thread, key-contiguous
  size_t i0 = (size_t)gid * 4;
  int bq = (int)(i0 >> 11);                   // (b*S + q)
  float inv = 1.0f / lsum[bq];
  float4* p = (float4*)attn_out + gid;
  float4 v = *p;
  v.x *= inv; v.y *= inv; v.z *= inv; v.w *= inv;
  *p = v;
}

// ---------------------------------------------------------------------------
// Kernel 4: output projection  out = concat @ W0 + b0 (M=4096,N=1024,K=1024)
// A tile (already bf16) staged by async DMA; W0 converted via packed cvt.
// ---------------------------------------------------------------------------
__global__ __launch_bounds__(256)
void out_proj_kernel(const bf16_t* __restrict__ Xc, const float* __restrict__ W0,
                     const float* __restrict__ b0, float* __restrict__ out) {
  __shared__ bf16_t As[128 * LD32];
  __shared__ bf16_t Bs[64 * LD32];
  const int tid  = threadIdx.x;
  const int wave = tid >> 5, lane = tid & 31;
  const int hi   = lane >> 4, lc = lane & 15;
  const int m0   = blockIdx.x * 128;
  const int n0   = blockIdx.y * 64;

  v8f acc[4] = {};
  for (int k0 = 0; k0 < D_; k0 += 32) {
    // stage A via async DMA: 128 rows x 32 bf16 = 512 x 16B chunks
#pragma unroll
    for (int i = 0; i < 2; ++i) {
      int idx = tid + i * 256;
      int r = idx >> 2, seg = idx & 3;
      async_b128(Xc + (size_t)(m0 + r) * D_ + k0 + seg * 8, &As[r * LD32 + seg * 8]);
    }
    // stage B transposed f32 -> bf16, packed pairs along k
#pragma unroll
    for (int i = 0; i < 4; ++i) {
      int idx = tid + i * 256;
      int n = idx & 63, kp = idx >> 6;
      const float* src = W0 + (size_t)(k0 + 2 * kp) * D_ + n0 + n;
      ((uint32_t*)Bs)[n * (LD32 / 2) + kp] = f2bf_pk(src[0], src[D_]);
    }
    wait_async0();
    __syncthreads();
    v16bf a = ld_frag_a16(&As[(wave * 16 + lc) * LD32 + hi * 16]);
#pragma unroll
    for (int nt = 0; nt < 4; ++nt) {
      v16bf b = ld_frag_b16(&Bs[(nt * 16 + lc) * LD32 + hi * 16]);
      acc[nt] = WMMA_BF16(a, b, acc[nt]);
    }
    __syncthreads();
  }
#pragma unroll
  for (int nt = 0; nt < 4; ++nt) {
#pragma unroll
    for (int r = 0; r < 8; ++r) {
      int row = wave * 16 + hi * 8 + r;
      int col = n0 + nt * 16 + lc;
      out[(size_t)(m0 + row) * D_ + col] = acc[nt][r] + b0[col];
    }
  }
}

// ---------------------------------------------------------------------------
extern "C" void kernel_launch(void* const* d_in, const int* in_sizes, int n_in,
                              void* d_out, int out_size, void* d_ws, size_t ws_size,
                              hipStream_t stream) {
  (void)in_sizes; (void)n_in; (void)out_size; (void)ws_size;
  const float* v    = (const float*)d_in[0];
  const float* k    = (const float*)d_in[1];
  const float* q    = (const float*)d_in[2];
  const int*   mask = (const int*)d_in[3];
  const float* Wq   = (const float*)d_in[4];
  const float* bq   = (const float*)d_in[5];
  const float* Wk   = (const float*)d_in[6];
  const float* bk   = (const float*)d_in[7];
  const float* Wv   = (const float*)d_in[8];
  const float* bv   = (const float*)d_in[9];
  const float* W0   = (const float*)d_in[10];
  const float* b0   = (const float*)d_in[11];

  float* out      = (float*)d_out;
  float* attn_out = out + (size_t)B_ * S_ * D_;   // [B,S,S] region of d_out

  // Workspace carve (~32 MB): qh/kh/vh/concat bf16 + last-head row sums f32
  const size_t head_elems = (size_t)B_ * H_ * S_ * DK_;   // 4,194,304
  bf16_t* qh     = (bf16_t*)d_ws;
  bf16_t* kh     = qh + head_elems;
  bf16_t* vh     = kh + head_elems;
  bf16_t* concat = vh + head_elems;                       // B*S*D bf16
  float*  lsum   = (float*)(concat + (size_t)B_ * S_ * D_);

  dim3 blk(256);
  qkv_proj_kernel<<<dim3(M_ / 128, H_), blk, 0, stream>>>(q, Wq, bq, qh);
  qkv_proj_kernel<<<dim3(M_ / 128, H_), blk, 0, stream>>>(k, Wk, bk, kh);
  qkv_proj_kernel<<<dim3(M_ / 128, H_), blk, 0, stream>>>(v, Wv, bv, vh);
  attn_kernel<<<dim3(S_ / 128, H_, B_), blk, 0, stream>>>(qh, kh, vh, mask,
                                                          concat, attn_out, lsum);
  attn_norm_kernel<<<dim3((B_ * S_ * S_ / 4) / 256), blk, 0, stream>>>(attn_out, lsum);
  out_proj_kernel<<<dim3(M_ / 128, D_ / 64), blk, 0, stream>>>(concat, W0, b0, out);
}
